// WavNet_17617955848652
// MI455X (gfx1250) — matrix operand order repaired and checked by reading, not compiled, not run on hardware
//
#include <hip/hip_runtime.h>

// ---------------------------------------------------------------------------
// WaveNet forward, MI455X (gfx1250). bf16 WMMA with f32 accumulation.
// Time-major activations (h[b][t][128], skip[b][t][256]); tap tiles staged
// by the Tensor Data Mover (tensor_load_to_lds w/ hardware LDS row padding),
// manual vectorized copy as fallback / for boundary tiles. Res+skip GEMMs
// share B-fragments; biases loaded as float4.
// ---------------------------------------------------------------------------

#define TMAX   8192
#define FINAL  5123
#define TOFF   3069
#define NT     64          // time tile per workgroup
#define RS1    136         // LDS row stride (halves) for 128-channel tiles
#define RS2    264         // LDS row stride (halves) for 256-channel tiles

typedef __attribute__((ext_vector_type(8)))  float  v8f;
typedef __attribute__((ext_vector_type(16))) __bf16 v16bf;
typedef __attribute__((ext_vector_type(2)))  __bf16 v2bf;
typedef __attribute__((ext_vector_type(4)))  unsigned int u32x4;
typedef __attribute__((ext_vector_type(4)))  int    i32x4;
typedef __attribute__((ext_vector_type(8)))  int    i32x8;

#if __has_builtin(__builtin_amdgcn_tensor_load_to_lds) && __has_builtin(__builtin_amdgcn_s_wait_tensorcnt)
#define HAVE_TDM 1
#else
#define HAVE_TDM 0
#endif

union Frag {
    uint4          q[2];
    v16bf          v;
    unsigned short h[16];
};
union U4 {
    uint4          q;
    unsigned int   u[4];
    unsigned short h[8];
};

__device__ __forceinline__ unsigned short f2bf(float f) {
    unsigned int u = __float_as_uint(f);
    u += 0x7FFFu + ((u >> 16) & 1u);
    return (unsigned short)(u >> 16);
}
__device__ __forceinline__ unsigned pk2bf(float a, float b) {
#if __has_builtin(__builtin_amdgcn_cvt_pk_bf16_f32)
    union { v2bf v; unsigned u; } c;
    c.v = __builtin_amdgcn_cvt_pk_bf16_f32(a, b);
    return c.u;
#else
    return (unsigned)f2bf(a) | ((unsigned)f2bf(b) << 16);
#endif
}
__device__ __forceinline__ float bf2f(unsigned short h) {
    return __uint_as_float(((unsigned int)h) << 16);
}
__device__ __forceinline__ float fast_tanh(float x) {
    float e = __expf(2.0f * x);
    return 1.0f - __fdividef(2.0f, e + 1.0f);
}
__device__ __forceinline__ float fast_sig(float x) {
    return __fdividef(1.0f, 1.0f + __expf(-x));
}
__device__ __forceinline__ v8f vzero() {
    v8f z;
#pragma unroll
    for (int i = 0; i < 8; ++i) z[i] = 0.0f;
    return z;
}
__device__ __forceinline__ v8f wmma_bf16(const Frag& a, const Frag& b, v8f c) {
    return __builtin_amdgcn_wmma_f32_16x16x32_bf16(
        false, a.v, false, b.v, (short)0, c, false, false);
}
__device__ __forceinline__ Frag ld_frag(const unsigned short* p) {
    Frag f;
    const uint4* q = (const uint4*)p;
    f.q[0] = q[0]; f.q[1] = q[1];
    return f;
}

#if HAVE_TDM
__device__ __forceinline__ unsigned lds_off_of(const void* p) {
    return (unsigned)(unsigned long long)
        (__attribute__((address_space(3))) const void*)p;
}
// DMA one 64-row x 128-half time-major tile into LDS, hardware row padding
// (64 DWORDs data + 4 DWORDs pad -> RS1 = 136-half rows).
__device__ __forceinline__ void tdm_load_tile(const unsigned short* gsrc,
                                              unsigned lds_byte_off) {
    unsigned long long ga = (unsigned long long)gsrc;
    u32x4 g0;
    g0[0] = 1u;                                   // count=1, user mode
    g0[1] = lds_byte_off;                         // lds_addr
    g0[2] = (unsigned)ga;                         // global_addr[31:0]
    g0[3] = (unsigned)((ga >> 32) & 0x01FFFFFFull) | (2u << 30); // type=2
    i32x8 g1;
    g1[0] = (1 << 16)        // data_size = 2B
          | (1 << 20)        // pad_enable
          | (5 << 22)        // pad_interval: 64 DWORDs
          | (3 << 25);       // pad_amount: 4 DWORDs
    g1[1] = 0;                                    // no atomic barrier
    g1[2] = (int)0x4000;                          // tensor_dim0 = 1<<30
    g1[3] = (int)(0x4000 | (128 << 16));          // tensor_dim1=1<<30, tile_dim0=128
    g1[4] = (int)NT;                              // tile_dim1 = 64 rows
    g1[5] = (int)128;                             // tensor_dim0_stride = 128
    g1[6] = 0;
    g1[7] = 0;
    i32x4 z4 = {0, 0, 0, 0};
#if __clang_major__ >= 23
    i32x8 z8 = {0, 0, 0, 0, 0, 0, 0, 0};
    __builtin_amdgcn_tensor_load_to_lds(g0, g1, z4, z4, z8, 0);
#else
    __builtin_amdgcn_tensor_load_to_lds(g0, g1, z4, z4, 0);
#endif
}
#endif

// ---------------------------------------------------------------------------
// Weight packing: fp32 [M x K] -> bf16 A-fragments, contiguous k = 16*(lane/16) + element.
// ---------------------------------------------------------------------------
__device__ __forceinline__ int klocal_of(int half, int lhi) {
    return lhi * 16 + half;
}

__global__ __launch_bounds__(256) void pack_simple(
    unsigned short* __restrict__ dst, const float* __restrict__ src,
    int M, int K, int srcStride)
{
    int l    = blockIdx.y;
    int idx  = blockIdx.x * 256 + threadIdx.x;
    int half = idx & 15;
    int lane = (idx >> 4) & 31;
    int frag = idx >> 9;
    int kt   = K >> 5;
    int mt   = frag / kt;
    int ks   = frag - mt * kt;
    int m    = mt * 16 + (lane & 15);
    int kk   = ks * 32 + klocal_of(half, lane >> 4);
    dst[(size_t)l * M * K + idx] =
        f2bf(src[(size_t)l * srcStride + (size_t)m * K + kk]);
}

__global__ __launch_bounds__(256) void pack_fg(
    unsigned short* __restrict__ dst,
    const float* __restrict__ fw, const float* __restrict__ gw)
{
    int l    = blockIdx.y;
    int idx  = blockIdx.x * 256 + threadIdx.x;
    int half = idx & 15;
    int lane = (idx >> 4) & 31;
    int frag = idx >> 9;
    int mt   = frag >> 3;
    int ks   = frag & 7;
    int m    = mt * 16 + (lane & 15);
    int kk   = ks * 32 + klocal_of(half, lane >> 4);
    int i    = kk & 127;
    int tap  = kk >> 7;
    const float* src = (m < 128) ? fw : gw;
    int o    = m & 127;
    float w  = src[(((size_t)l * 128 + o) * 128 + i) * 2 + tap];
    dst[(size_t)l * 65536 + idx] = f2bf(w);
}

// ---------------------------------------------------------------------------
// Start conv: h0[b][t][128] = start_w[128x256] * x[b][256][t]
// ---------------------------------------------------------------------------
__global__ __launch_bounds__(256) void wav_start(
    const float* __restrict__ x, unsigned short* __restrict__ h0,
    const unsigned short* __restrict__ wS)
{
    __shared__ __align__(16) unsigned short xs[NT * RS2];

    const int tid = threadIdx.x;
    const int b   = blockIdx.y;
    const int t0  = blockIdx.x * NT;

    {   // stage x tile (input is channel-major), transpose + bf16
        int c = tid;
        const float* row = x + ((size_t)b * 256 + c) * TMAX;
#pragma unroll 4
        for (int tt = 0; tt < NT; ++tt) {
            xs[tt * RS2 + c] = f2bf(row[t0 + tt]);
        }
    }
    __syncthreads();

    const int wv = tid >> 5, lane = tid & 31;
    const int lhi = lane >> 4, ln = lane & 15;

    v8f acc[4];
#pragma unroll
    for (int n = 0; n < 4; ++n) acc[n] = vzero();

    for (int ks = 0; ks < 8; ++ks) {
        Frag a = ld_frag(wS + ((size_t)(wv * 8 + ks)) * 512 + lane * 16);
        int c0 = ks * 32 + lhi * 16;
#pragma unroll
        for (int n = 0; n < 4; ++n) {
            Frag bb = ld_frag(xs + (n * 16 + ln) * RS2 + c0);
            acc[n] = wmma_bf16(a, bb, acc[n]);
        }
    }

    unsigned short* hb = h0 + (size_t)b * TMAX * 128;
    const int mb = wv * 16 + lhi * 8;
#pragma unroll
    for (int n = 0; n < 4; ++n) {
        int t = t0 + n * 16 + ln;
        U4 pk;
#pragma unroll
        for (int r = 0; r < 8; r += 2)
            pk.u[r >> 1] = pk2bf(acc[n][r], acc[n][r + 1]);
        *(uint4*)(hb + (size_t)t * 128 + mb) = pk.q;
    }
}

// ---------------------------------------------------------------------------
// Fused WaveNet layer (time-major h):
//   new_h[t] = res_w * (tanh(.)*sig(.)) + res_b + h[t],  t >= t_start
//   skipAcc[t] += skip_w * gated + skip_b,               t >= TOFF
// ---------------------------------------------------------------------------
__global__ __launch_bounds__(256) void wav_layer(
    const unsigned short* __restrict__ hIn,   // [b][t][128]
    unsigned short* __restrict__ hOut,        // [b][t][128]
    const unsigned short* __restrict__ wFG,
    const unsigned short* __restrict__ wRes,
    const unsigned short* __restrict__ wSkip,
    const float* __restrict__ resB,
    const float* __restrict__ skipB,
    float* __restrict__ skipAcc,              // [b][tf][256]
    int d, int t_start, int TBASE)
{
    __shared__ __align__(16) unsigned short tap0[NT * RS1];
    __shared__ __align__(16) unsigned short tap1[NT * RS1];
    __shared__ __align__(16) unsigned short gated[NT * RS1];

    const int tid = threadIdx.x;
    const int b   = blockIdx.y;
    const int t0  = TBASE + blockIdx.x * NT;
    const unsigned short* hInB = hIn + (size_t)b * TMAX * 128;

    const bool interior = (t0 - d >= 0) && (t0 + NT <= TMAX);
#if HAVE_TDM
    if (interior) {
        if (tid < 32) {     // wave 0 drives the DMA
            tdm_load_tile(hInB + (size_t)(t0 - d) * 128, lds_off_of(tap0));
            tdm_load_tile(hInB + (size_t)t0 * 128,       lds_off_of(tap1));
            __builtin_amdgcn_s_wait_tensorcnt(0);
        }
    } else
#endif
    {   // manual vectorized staging (fallback / boundary tiles): 16B chunks
        (void)interior;
#pragma unroll
        for (int k = 0; k < 8; ++k) {
            int ch  = k * 256 + tid;
            int tap = ch >> 10;
            int rr  = (ch >> 4) & 63;
            int cc  = ch & 15;
            int t   = t0 + rr - (tap ? 0 : d);
            if (t < 0) t = 0;
            if (t > TMAX - 1) t = TMAX - 1;
            uint4 v = *(const uint4*)(hInB + (size_t)t * 128 + cc * 8);
            *(uint4*)((tap ? tap1 : tap0) + rr * RS1 + cc * 8) = v;
        }
    }
    __syncthreads();

    const int wv = tid >> 5, lane = tid & 31;
    const int lhi = lane >> 4, ln = lane & 15;
    const int mb = wv * 16 + lhi * 8;

    // ---- GEMM1: fused filter(rows wv) / gate(rows wv+8), K=256 (2 taps)
    v8f accF[4], accG[4];
#pragma unroll
    for (int n = 0; n < 4; ++n) { accF[n] = vzero(); accG[n] = vzero(); }

    for (int ks = 0; ks < 8; ++ks) {
        Frag af = ld_frag(wFG + ((size_t)(wv * 8 + ks)) * 512 + lane * 16);
        Frag ag = ld_frag(wFG + ((size_t)((wv + 8) * 8 + ks)) * 512 + lane * 16);
        const unsigned short* tp = (ks < 4) ? tap0 : tap1;
        int c0 = (ks & 3) * 32 + lhi * 16;
#pragma unroll
        for (int n = 0; n < 4; ++n) {
            Frag bb = ld_frag(tp + (n * 16 + ln) * RS1 + c0);
            accF[n] = wmma_bf16(af, bb, accF[n]);
            accG[n] = wmma_bf16(ag, bb, accG[n]);
        }
    }

    // gated activation -> LDS [t][c], packed b128 store
#pragma unroll
    for (int n = 0; n < 4; ++n) {
        int tt = n * 16 + ln;
        U4 pk;
#pragma unroll
        for (int r = 0; r < 8; r += 2) {
            float v0 = fast_tanh(accF[n][r])     * fast_sig(accG[n][r]);
            float v1 = fast_tanh(accF[n][r + 1]) * fast_sig(accG[n][r + 1]);
            pk.u[r >> 1] = pk2bf(v0, v1);
        }
        *(uint4*)(gated + tt * RS1 + mb) = pk.q;
    }
    __syncthreads();

    // ---- GEMM2+3 merged: res(row wv), skip(rows wv, wv+8) share B-frags
    v8f accR[4], accS[2][4];
#pragma unroll
    for (int n = 0; n < 4; ++n) {
        accR[n] = vzero(); accS[0][n] = vzero(); accS[1][n] = vzero();
    }

    for (int ks = 0; ks < 4; ++ks) {
        Frag aR  = ld_frag(wRes  + ((size_t)(wv * 4 + ks)) * 512 + lane * 16);
        Frag aS0 = ld_frag(wSkip + ((size_t)(wv * 4 + ks)) * 512 + lane * 16);
        Frag aS1 = ld_frag(wSkip + ((size_t)((wv + 8) * 4 + ks)) * 512 + lane * 16);
        int c0 = ks * 32 + lhi * 16;
#pragma unroll
        for (int n = 0; n < 4; ++n) {
            Frag bb = ld_frag(gated + (n * 16 + ln) * RS1 + c0);
            accR[n]    = wmma_bf16(aR,  bb, accR[n]);
            accS[0][n] = wmma_bf16(aS0, bb, accS[0][n]);
            accS[1][n] = wmma_bf16(aS1, bb, accS[1][n]);
        }
    }

    // residual epilogue -> hOut (bf16, packed b128)
    unsigned short* hOutB = hOut + (size_t)b * TMAX * 128;
    const float4 rb0 = *(const float4*)(resB + mb);
    const float4 rb1 = *(const float4*)(resB + mb + 4);
#pragma unroll
    for (int n = 0; n < 4; ++n) {
        int tt = n * 16 + ln;
        int t  = t0 + tt;
        U4 hv; hv.q = *(const uint4*)(tap1 + tt * RS1 + mb);
        if (t >= t_start && t < TMAX) {
            U4 pk;
            pk.u[0] = pk2bf(accR[n][0] + rb0.x + bf2f(hv.h[0]),
                            accR[n][1] + rb0.y + bf2f(hv.h[1]));
            pk.u[1] = pk2bf(accR[n][2] + rb0.z + bf2f(hv.h[2]),
                            accR[n][3] + rb0.w + bf2f(hv.h[3]));
            pk.u[2] = pk2bf(accR[n][4] + rb1.x + bf2f(hv.h[4]),
                            accR[n][5] + rb1.y + bf2f(hv.h[5]));
            pk.u[3] = pk2bf(accR[n][6] + rb1.z + bf2f(hv.h[6]),
                            accR[n][7] + rb1.w + bf2f(hv.h[7]));
            *(uint4*)(hOutB + (size_t)t * 128 + mb) = pk.q;
        }
    }

    // skip epilogue -> skipAcc (f32 time-major, contiguous float4 RMW)
#pragma unroll
    for (int p = 0; p < 2; ++p) {
        int ms = (wv + p * 8) * 16 + lhi * 8;
        const float4 sb0 = *(const float4*)(skipB + ms);
        const float4 sb1 = *(const float4*)(skipB + ms + 4);
#pragma unroll
        for (int n = 0; n < 4; ++n) {
            int t = t0 + n * 16 + ln;
            if (t >= TOFF && t < TMAX) {
                float* sp = skipAcc + ((size_t)b * FINAL + (t - TOFF)) * 256 + ms;
                float4 v0 = *(float4*)(sp);
                float4 v1 = *(float4*)(sp + 4);
                v0.x += accS[p][n][0] + sb0.x;
                v0.y += accS[p][n][1] + sb0.y;
                v0.z += accS[p][n][2] + sb0.z;
                v0.w += accS[p][n][3] + sb0.w;
                v1.x += accS[p][n][4] + sb1.x;
                v1.y += accS[p][n][5] + sb1.y;
                v1.z += accS[p][n][6] + sb1.z;
                v1.w += accS[p][n][7] + sb1.w;
                *(float4*)(sp)     = v0;
                *(float4*)(sp + 4) = v1;
            }
        }
    }
}

// ---------------------------------------------------------------------------
// Post: out = post2 * relu(post1 * relu(skip) + b1) + b2
// ---------------------------------------------------------------------------
__global__ __launch_bounds__(256) void wav_post(
    const float* __restrict__ skipAcc,        // [b][tf][256]
    const unsigned short* __restrict__ wP1,
    const unsigned short* __restrict__ wP2,
    const float* __restrict__ b1, const float* __restrict__ b2,
    float* __restrict__ out)                  // [b][256][FINAL]
{
    __shared__ __align__(16) unsigned short zs[NT * RS2];
    __shared__ __align__(16) unsigned short es[NT * RS2];

    const int tid = threadIdx.x;
    const int b   = blockIdx.y;
    const int t0  = blockIdx.x * NT;

    {   // stage relu(skip): contiguous float4 rows -> bf16 LDS rows
#pragma unroll
        for (int k = 0; k < 16; ++k) {
            int id = k * 256 + tid;
            int rr = id >> 6;
            int cc = id & 63;
            int tf = t0 + rr;
            float4 v;
            if (tf < FINAL) {
                v = *(const float4*)(skipAcc + ((size_t)b * FINAL + tf) * 256 + cc * 4);
            } else {
                v.x = v.y = v.z = v.w = 0.0f;
            }
            v.x = v.x > 0.0f ? v.x : 0.0f;
            v.y = v.y > 0.0f ? v.y : 0.0f;
            v.z = v.z > 0.0f ? v.z : 0.0f;
            v.w = v.w > 0.0f ? v.w : 0.0f;
            uint2 pp;
            pp.x = pk2bf(v.x, v.y);
            pp.y = pk2bf(v.z, v.w);
            *(uint2*)(zs + rr * RS2 + cc * 4) = pp;
        }
    }
    __syncthreads();

    const int wv = tid >> 5, lane = tid & 31;
    const int lhi = lane >> 4, ln = lane & 15;

    // post1
    v8f a1c[2][4];
#pragma unroll
    for (int p = 0; p < 2; ++p)
#pragma unroll
        for (int n = 0; n < 4; ++n) a1c[p][n] = vzero();

    for (int ks = 0; ks < 8; ++ks) {
        Frag f0 = ld_frag(wP1 + ((size_t)(wv * 8 + ks)) * 512 + lane * 16);
        Frag f1 = ld_frag(wP1 + ((size_t)((wv + 8) * 8 + ks)) * 512 + lane * 16);
        int c0 = ks * 32 + lhi * 16;
#pragma unroll
        for (int n = 0; n < 4; ++n) {
            Frag bb = ld_frag(zs + (n * 16 + ln) * RS2 + c0);
            a1c[0][n] = wmma_bf16(f0, bb, a1c[0][n]);
            a1c[1][n] = wmma_bf16(f1, bb, a1c[1][n]);
        }
    }
#pragma unroll
    for (int p = 0; p < 2; ++p) {
        int ms = (wv + p * 8) * 16 + lhi * 8;
        const float4 c0v = *(const float4*)(b1 + ms);
        const float4 c1v = *(const float4*)(b1 + ms + 4);
#pragma unroll
        for (int n = 0; n < 4; ++n) {
            int tt = n * 16 + ln;
            float w0 = a1c[p][n][0] + c0v.x, w1 = a1c[p][n][1] + c0v.y;
            float w2 = a1c[p][n][2] + c0v.z, w3 = a1c[p][n][3] + c0v.w;
            float w4 = a1c[p][n][4] + c1v.x, w5 = a1c[p][n][5] + c1v.y;
            float w6 = a1c[p][n][6] + c1v.z, w7 = a1c[p][n][7] + c1v.w;
            U4 pk;
            pk.u[0] = pk2bf(w0 > 0.f ? w0 : 0.f, w1 > 0.f ? w1 : 0.f);
            pk.u[1] = pk2bf(w2 > 0.f ? w2 : 0.f, w3 > 0.f ? w3 : 0.f);
            pk.u[2] = pk2bf(w4 > 0.f ? w4 : 0.f, w5 > 0.f ? w5 : 0.f);
            pk.u[3] = pk2bf(w6 > 0.f ? w6 : 0.f, w7 > 0.f ? w7 : 0.f);
            *(uint4*)(es + tt * RS2 + ms) = pk.q;
        }
    }
    __syncthreads();

    // post2
    v8f a2c[2][4];
#pragma unroll
    for (int p = 0; p < 2; ++p)
#pragma unroll
        for (int n = 0; n < 4; ++n) a2c[p][n] = vzero();

    for (int ks = 0; ks < 8; ++ks) {
        Frag f0 = ld_frag(wP2 + ((size_t)(wv * 8 + ks)) * 512 + lane * 16);
        Frag f1 = ld_frag(wP2 + ((size_t)((wv + 8) * 8 + ks)) * 512 + lane * 16);
        int c0 = ks * 32 + lhi * 16;
#pragma unroll
        for (int n = 0; n < 4; ++n) {
            Frag bb = ld_frag(es + (n * 16 + ln) * RS2 + c0);
            a2c[0][n] = wmma_bf16(f0, bb, a2c[0][n]);
            a2c[1][n] = wmma_bf16(f1, bb, a2c[1][n]);
        }
    }
#pragma unroll
    for (int p = 0; p < 2; ++p) {
        int ms = (wv + p * 8) * 16 + lhi * 8;
        const float4 c0v = *(const float4*)(b2 + ms);
        const float4 c1v = *(const float4*)(b2 + ms + 4);
#pragma unroll
        for (int n = 0; n < 4; ++n) {
            int tf = t0 + n * 16 + ln;
            if (tf < FINAL) {
                float* op = out + (size_t)b * 256 * FINAL + tf;
                op[(size_t)(ms + 0) * FINAL] = a2c[p][n][0] + c0v.x;
                op[(size_t)(ms + 1) * FINAL] = a2c[p][n][1] + c0v.y;
                op[(size_t)(ms + 2) * FINAL] = a2c[p][n][2] + c0v.z;
                op[(size_t)(ms + 3) * FINAL] = a2c[p][n][3] + c0v.w;
                op[(size_t)(ms + 4) * FINAL] = a2c[p][n][4] + c1v.x;
                op[(size_t)(ms + 5) * FINAL] = a2c[p][n][5] + c1v.y;
                op[(size_t)(ms + 6) * FINAL] = a2c[p][n][6] + c1v.z;
                op[(size_t)(ms + 7) * FINAL] = a2c[p][n][7] + c1v.w;
            }
        }
    }
}

// ---------------------------------------------------------------------------
// Host launcher
// ---------------------------------------------------------------------------
extern "C" void kernel_launch(void* const* d_in, const int* in_sizes, int n_in,
                              void* d_out, int out_size, void* d_ws, size_t ws_size,
                              hipStream_t stream)
{
    (void)in_sizes; (void)n_in; (void)out_size; (void)ws_size;

    const float* x       = (const float*)d_in[0];
    const float* start_w = (const float*)d_in[1];
    const float* filt_w  = (const float*)d_in[2];
    const float* gate_w  = (const float*)d_in[3];
    const float* res_w   = (const float*)d_in[4];
    const float* res_b   = (const float*)d_in[5];
    const float* skip_w  = (const float*)d_in[6];
    const float* skip_b  = (const float*)d_in[7];
    const float* p1_w    = (const float*)d_in[8];
    const float* p1_b    = (const float*)d_in[9];
    const float* p2_w    = (const float*)d_in[10];
    const float* p2_b    = (const float*)d_in[11];
    float* out           = (float*)d_out;

    size_t off = 0;
    auto take = [&](size_t bytes) {
        void* p = (char*)d_ws + off;
        off = (off + bytes + 255) & ~(size_t)255;
        return p;
    };
    const size_t szH    = (size_t)8 * TMAX * 128 * 2;
    const size_t szSkip = (size_t)8 * FINAL * 256 * 4;
    unsigned short* hA    = (unsigned short*)take(szH);
    unsigned short* hB    = (unsigned short*)take(szH);
    float*          skipA = (float*)take(szSkip);
    unsigned short* wFG   = (unsigned short*)take((size_t)30 * 65536 * 2);
    unsigned short* wRes  = (unsigned short*)take((size_t)30 * 16384 * 2);
    unsigned short* wSk   = (unsigned short*)take((size_t)30 * 32768 * 2);
    unsigned short* wSt   = (unsigned short*)take((size_t)32768 * 2);
    unsigned short* wP1   = (unsigned short*)take((size_t)65536 * 2);
    unsigned short* wP2   = (unsigned short*)take((size_t)65536 * 2);

    hipMemsetAsync(skipA, 0, szSkip, stream);

    pack_fg     <<<dim3(256, 30), 256, 0, stream>>>(wFG, filt_w, gate_w);
    pack_simple <<<dim3( 64, 30), 256, 0, stream>>>(wRes, res_w, 128, 128, 16384);
    pack_simple <<<dim3(128, 30), 256, 0, stream>>>(wSk,  skip_w, 256, 128, 32768);
    pack_simple <<<dim3(128,  1), 256, 0, stream>>>(wSt,  start_w, 128, 256, 0);
    pack_simple <<<dim3(256,  1), 256, 0, stream>>>(wP1,  p1_w, 256, 256, 0);
    pack_simple <<<dim3(256,  1), 256, 0, stream>>>(wP2,  p2_w, 256, 256, 0);

    wav_start<<<dim3(TMAX / NT, 8), 256, 0, stream>>>(x, hA, wSt);

    static const int DIL[10] = {1, 2, 4, 8, 16, 32, 64, 128, 256, 512};
    int offT = 0;
    for (int l = 0; l < 30; ++l) {
        int d = DIL[l % 10];
        int t_start = offT + d;
        int TBASE   = (t_start / NT) * NT;
        int ntiles  = (TMAX - TBASE + NT - 1) / NT;
        unsigned short* hin  = (l & 1) ? hB : hA;
        unsigned short* hout = (l & 1) ? hA : hB;
        wav_layer<<<dim3(ntiles, 8), 256, 0, stream>>>(
            hin, hout,
            wFG + (size_t)l * 65536,
            wRes + (size_t)l * 16384,
            wSk + (size_t)l * 32768,
            res_b + (size_t)l * 128,
            skip_b + (size_t)l * 256,
            skipA, d, t_start, TBASE);
        offT = t_start;
    }

    wav_post<<<dim3((FINAL + NT - 1) / NT, 8), 256, 0, stream>>>(
        skipA, wP1, wP2, p1_b, p2_b, out);
}
